// MMD_29729763623080
// MI455X (gfx1250) — matrix operand order repaired and checked by reading, not compiled, not run on hardware
//
#include <hip/hip_runtime.h>
#include <hip/hip_bf16.h>

typedef __attribute__((ext_vector_type(16))) _Float16 v16h;
typedef __attribute__((ext_vector_type(8)))  float    v8f;

#define B_    2
#define Q_    75
#define NWAY  5
#define KSHOT 5
#define S_    25      // NWAY*KSHOT
#define C_    640
#define P_    25      // H*W
#define NH_   20
#define IMG_  (C_*P_) // 16000 floats per image
#define NROW  125     // KSHOT*P_
#define TPB2  768     // 24 waves: wave = gram(3) x row-strip(8)
#define INV_SCALE 0.17677669529663687f  // 1/sqrt(32)

// multi-scale gaussian kernel: alphas = 2^k, k in [-3,2) -> exp chain via squaring
__device__ __forceinline__ float mgk5(float d2) {
    d2 = fmaxf(d2, 0.0f);
    float e1 = __expf(-0.125f * d2);
    float e2 = e1 * e1, e4 = e2 * e2, e8 = e4 * e4, e16 = e8 * e8;
    return e1 + e2 + e4 + e8 + e16;
}

// ---------------- Kernel 1: QKV projections (1x1 conv == channel matmul) ----
__global__ __launch_bounds__(256) void proj_kernel(
    const float* __restrict__ sup, const float* __restrict__ qry,
    const float* __restrict__ Wq, const float* __restrict__ Wk, const float* __restrict__ Wv,
    float* __restrict__ Psq, float* __restrict__ Psk, float* __restrict__ Psv,
    float* __restrict__ Pqq, float* __restrict__ Pqk, float* __restrict__ Pqv)
{
    __shared__ float xs[IMG_];   // 64000 B
    int img = blockIdx.x;        // 0..199
    int b = img / 100, r = img % 100;
    const float* src;
    float *dq, *dk, *dv;
    size_t obase;
    if (r < S_) {
        src = sup + (size_t)(b * S_ + r) * IMG_;
        obase = (size_t)(b * S_ + r) * IMG_;
        dq = Psq; dk = Psk; dv = Psv;
    } else {
        int n = r - S_;
        src = qry + (size_t)(b * Q_ + n) * IMG_;
        obase = (size_t)(b * Q_ + n) * IMG_;
        dq = Pqq; dk = Pqk; dv = Pqv;
    }
    for (int idx = threadIdx.x; idx < IMG_; idx += 256) xs[idx] = src[idx];
    __syncthreads();
    for (int oi = threadIdx.x; oi < IMG_; oi += 256) {
        int o = oi / P_, p = oi % P_;
        const float* wq = Wq + (size_t)o * C_;
        const float* wk = Wk + (size_t)o * C_;
        const float* wv = Wv + (size_t)o * C_;
        float aq = 0.f, ak = 0.f, av = 0.f;
        #pragma unroll 4
        for (int c = 0; c < C_; ++c) {
            float xv = xs[c * P_ + p];
            aq = fmaf(wq[c], xv, aq);
            ak = fmaf(wk[c], xv, ak);
            av = fmaf(wv[c], xv, av);
        }
        dq[obase + oi] = aq;
        dk[obase + oi] = ak;
        dv[obase + oi] = av;
    }
}

// ---------------- Kernel 2: fused cross-attention + WMMA Grams + MMD --------
// One block per (b, q, way) group. 768 threads = 24 waves.
// Wave w: gram = w/8 (0:SS 1:QQ 2:QS), row-strip rt = w%8 -> 8 accum fragments.
__global__ __launch_bounds__(TPB2) void group_kernel(
    const float* __restrict__ sup, const float* __restrict__ qry,
    const float* __restrict__ Psq, const float* __restrict__ Psk, const float* __restrict__ Psv,
    const float* __restrict__ Pqq, const float* __restrict__ Pqk, const float* __restrict__ Pqv,
    float* __restrict__ mmd_out)
{
    __shared__ _Float16 stQ[3 * P_ * 32];        // query-image q/k^T/v  [which][pos][dim]
    __shared__ _Float16 stS[KSHOT * 3 * P_ * 32];// per-shot  q/k^T/v
    __shared__ _Float16 sxfh[128][34];           // per-head S features (f16, padded)
    __shared__ _Float16 qxfh[128][34];           // per-head Q features
    __shared__ float rnS[128], mS[128], rnQ[128], mQ[128];
    __shared__ float wred[24];

    const int tid  = threadIdx.x;
    const int grp  = blockIdx.x;          // ((b*Q)+q)*NWAY + way
    const int way  = grp % NWAY;
    const int q    = (grp / NWAY) % Q_;
    const int b    = grp / (NWAY * Q_);

    // zero pad rows 125..127 once (never written afterwards)
    if (tid < 3 * 34) {
        int rr = NROW + tid / 34, cc = tid % 34;
        sxfh[rr][cc] = (_Float16)0.0f;
        qxfh[rr][cc] = (_Float16)0.0f;
    }

    const int lane = tid & 31, wvid = tid >> 5;
    const int gram = wvid >> 3;           // 0:SS 1:QQ 2:QS
    const int rt   = wvid & 7;            // row-strip
    const int mA   = lane & 15, gA = lane >> 4;

    // LDS sources for this wave's gram (select of two addrspace(3) ptrs stays DS)
    _Float16 (* __restrict__ Asrc)[34] = (gram == 1) ? qxfh : sxfh;  // rows
    _Float16 (* __restrict__ Bsrc)[34] = (gram == 0) ? sxfh : qxfh;  // cols

    v8f acc[8];
    v8f zero = {};
    #pragma unroll
    for (int ct = 0; ct < 8; ++ct) acc[ct] = zero;

    float rowsum = 0.f, rownorm = 0.f;   // per-thread row stats across heads

    for (int h = 0; h < NH_; ++h) {
        __syncthreads();   // prior head's WMMA fragment loads done
        // ---- stage per-head QKV into LDS (f16) ----
        for (int idx = tid; idx < 14400; idx += TPB2) {
            if (idx < 2400) {
                int which = idx / 800, r = idx % 800;
                int d = r / P_, pos = r % P_;
                const float* Pb = (which == 0) ? Pqq : (which == 1) ? Pqk : Pqv;
                float v = Pb[((size_t)(b * Q_ + q) * C_ + (h * 32 + d)) * P_ + pos];
                stQ[(which * P_ + pos) * 32 + d] = (_Float16)v;
            } else {
                int j = idx - 2400;
                int shot = j / 2400, r2 = j % 2400;
                int which = r2 / 800, rr = r2 % 800;
                int d = rr / P_, pos = rr % P_;
                const float* Pb = (which == 0) ? Psq : (which == 1) ? Psk : Psv;
                int sg = way * KSHOT + shot;
                float v = Pb[((size_t)(b * S_ + sg) * C_ + (h * 32 + d)) * P_ + pos];
                stS[((shot * 3 + which) * P_ + pos) * 32 + d] = (_Float16)v;
            }
        }
        __syncthreads();
        // ---- attention: 250 tasks = shot(5) x dir(2) x row(25), waves 0..7 ----
        if (tid < 250) {
            int s = tid / 50, rem = tid % 50, dir = rem / 25, i = rem % 25;
            const _Float16 *qrow, *kT, *vT;
            if (dir == 0) {   // sxf: query's queries vs support s keys/values
                qrow = &stQ[(0 * P_ + i) * 32];
                kT   = &stS[((s * 3 + 1) * P_) * 32];
                vT   = &stS[((s * 3 + 2) * P_) * 32];
            } else {          // qxf: support s queries vs query's keys/values
                qrow = &stS[((s * 3 + 0) * P_ + i) * 32];
                kT   = &stQ[(1 * P_) * 32];
                vT   = &stQ[(2 * P_) * 32];
            }
            float l[25];
            float mx = -1e30f;
            for (int j = 0; j < 25; ++j) {
                float a = 0.f;
                #pragma unroll
                for (int d = 0; d < 32; ++d)
                    a = fmaf((float)qrow[d], (float)kT[j * 32 + d], a);
                a *= INV_SCALE;
                l[j] = a;
                mx = fmaxf(mx, a);
            }
            float ssum = 0.f;
            for (int j = 0; j < 25; ++j) { l[j] = __expf(l[j] - mx); ssum += l[j]; }
            float inv = 1.f / ssum;
            int row = s * 25 + i;
            size_t xbase;
            const float* xsrc;
            if (dir == 0) {
                xbase = ((size_t)(b * S_ + way * KSHOT + s) * C_ + h * 32) * P_ + i;
                xsrc = sup;
            } else {
                xbase = ((size_t)(b * Q_ + q) * C_ + h * 32) * P_ + i;
                xsrc = qry;
            }
            #pragma unroll 4
            for (int ch = 0; ch < 32; ++ch) {
                float a = 0.f;
                for (int j = 0; j < 25; ++j)
                    a = fmaf(l[j], (float)vT[j * 32 + ch], a);
                float val = a * inv + xsrc[xbase + (size_t)ch * P_];
                if (dir == 0) sxfh[row][ch] = (_Float16)val;
                else          qxfh[row][ch] = (_Float16)val;
                rowsum += val;
                rownorm = fmaf(val, val, rownorm);
            }
        }
        __syncthreads();
        // ---- WMMA Gram accumulation for this head (K = 32), 8 issues/wave ----
        {
            v16h A;
            #pragma unroll
            for (int hh = 0; hh < 16; ++hh) {   // A 16-bit 16x32 lane layout (ISA 7.12.2)
                int k = (hh & 7) + ((hh >> 3) << 4) + gA * 8;
                A[hh] = Asrc[rt * 16 + mA][k];
            }
            #pragma unroll
            for (int ct = 0; ct < 8; ++ct) {
                v16h Bf;
                #pragma unroll
                for (int hh = 0; hh < 16; ++hh) {  // B 16-bit 32x16: n=lane&15, k=(hh&7)+16*(hh>>3)+8*(lane>>4)
                    int k = (hh & 7) + ((hh >> 3) << 4) + gA * 8;
                    Bf[hh] = Bsrc[ct * 16 + mA][k];
                }
                acc[ct] = __builtin_amdgcn_wmma_f32_16x16x32_f16(
                    false, A, false, Bf, (short)0, acc[ct], false, false);
            }
        }
    }

    // ---- publish row stats (norms + means of uncentered rows) ----
    if (tid < 250) {
        int s = tid / 50, rem = tid % 50, dir = rem / 25, i = rem % 25;
        int row = s * 25 + i;
        if (dir == 0) { rnS[row] = rownorm; mS[row] = rowsum * (1.f / 640.f); }
        else          { rnQ[row] = rownorm; mQ[row] = rowsum * (1.f / 640.f); }
    }
    __syncthreads();

    // ---- MMD reduction: d2_centered = ri + rj - 2G - C*(mi-mj)^2 ----
    const float* rA = (gram == 1) ? rnQ : rnS;
    const float* mA_ = (gram == 1) ? mQ : mS;
    const float* rB = (gram == 0) ? rnS : rnQ;
    const float* mB_ = (gram == 0) ? mS : mQ;
    const bool skip_diag = (gram != 2);

    float ssum = 0.f;
    {
        int nn = lane & 15, g = lane >> 4;
        #pragma unroll
        for (int ct = 0; ct < 8; ++ct) {
            #pragma unroll
            for (int v = 0; v < 8; ++v) {
                int i = rt * 16 + v + 8 * g;   // C/D f32 layout: row = vgpr + 8*(lane>=16)
                int j = ct * 16 + nn;
                if (i < NROW && j < NROW && !(skip_diag && i == j)) {
                    float dm = mA_[i] - mB_[j];
                    float d2 = rA[i] + rB[j] - 2.f * acc[ct][v] - 640.f * dm * dm;
                    ssum += mgk5(d2);
                }
            }
        }
    }
    #pragma unroll
    for (int off = 16; off > 0; off >>= 1) ssum += __shfl_down(ssum, off, 32);
    if (lane == 0) wred[wvid] = ssum;     // wred[gram*8 + rt]
    __syncthreads();
    if (tid == 0) {
        float tss = 0.f, tqq = 0.f, tqs = 0.f;
        for (int w = 0; w < 8; ++w) {
            tss += wred[w];
            tqq += wred[8 + w];
            tqs += wred[16 + w];
        }
        const float inv1 = 1.f / (125.f * 124.f);
        mmd_out[grp] = tss * inv1 + tqq * inv1 - (2.f / (125.f * 125.f)) * tqs;
    }
}

// ---------------- Kernel 3: log-softmax NLL over 150 rows x 5 ways ----------
__global__ __launch_bounds__(256) void loss_kernel(
    const float* __restrict__ mmd, const int* __restrict__ qy, float* __restrict__ out)
{
    __shared__ float red[8];
    int tid = threadIdx.x;
    float contrib = 0.f;
    if (tid < B_ * Q_) {
        const float* m = mmd + tid * NWAY;
        float lg[NWAY], mx = -1e30f;
        #pragma unroll
        for (int w = 0; w < NWAY; ++w) { lg[w] = -m[w] * (1.f / 12.5f); mx = fmaxf(mx, lg[w]); }
        float s = 0.f;
        #pragma unroll
        for (int w = 0; w < NWAY; ++w) s += __expf(lg[w] - mx);
        float lse = mx + __logf(s);
        int y = qy[tid];
        contrib = -(lg[y] - lse);
    }
    #pragma unroll
    for (int off = 16; off > 0; off >>= 1) contrib += __shfl_down(contrib, off, 32);
    if ((tid & 31) == 0) red[tid >> 5] = contrib;
    __syncthreads();
    if (tid == 0) {
        float t = 0.f;
        for (int w = 0; w < 8; ++w) t += red[w];
        out[0] = t / (float)(B_ * Q_);
    }
}

extern "C" void kernel_launch(void* const* d_in, const int* in_sizes, int n_in,
                              void* d_out, int out_size, void* d_ws, size_t ws_size,
                              hipStream_t stream) {
    (void)in_sizes; (void)n_in; (void)out_size; (void)ws_size;
    const float* sup = (const float*)d_in[0];   // support_xf [2,25,640,5,5]
    const float* qry = (const float*)d_in[2];   // query_xf   [2,75,640,5,5]
    const int*   qy  = (const int*)  d_in[3];   // query_y    [2,75]
    const float* Wq  = (const float*)d_in[4];
    const float* Wk  = (const float*)d_in[5];
    const float* Wv  = (const float*)d_in[6];

    float* ws  = (float*)d_ws;
    float* Psq = ws;                 // 2*25*16000 = 800000
    float* Psk = Psq + 800000;
    float* Psv = Psk + 800000;
    float* Pqq = Psv + 800000;       // 2*75*16000 = 2400000 each
    float* Pqk = Pqq + 2400000;
    float* Pqv = Pqk + 2400000;
    float* mmd = Pqv + 2400000;      // 750

    proj_kernel<<<B_ * (S_ + Q_), 256, 0, stream>>>(sup, qry, Wq, Wk, Wv,
                                                    Psq, Psk, Psv, Pqq, Pqk, Pqv);
    group_kernel<<<B_ * Q_ * NWAY, TPB2, 0, stream>>>(sup, qry,
                                                      Psq, Psk, Psv, Pqq, Pqk, Pqv, mmd);
    loss_kernel<<<1, 256, 0, stream>>>(mmd, qy, (float*)d_out);
}
